// MultiHead_63221918597522
// MI455X (gfx1250) — matrix operand order repaired and checked by reading, not compiled, not run on hardware
//
#include <hip/hip_runtime.h>
#include <cmath>

typedef __bf16 bf16_t;
typedef __attribute__((ext_vector_type(16))) __bf16 v16bf;
typedef __attribute__((ext_vector_type(8)))  __bf16 v8bf;
typedef __attribute__((ext_vector_type(8)))  float   v8f;

// ---------------------------------------------------------------------------
// helpers
// ---------------------------------------------------------------------------
__device__ __forceinline__ v16bf combine16(v8bf lo, v8bf hi) {
  v16bf r;
#pragma unroll
  for (int i = 0; i < 8; ++i) { r[i] = lo[i]; r[i + 8] = hi[i]; }
  return r;
}

__device__ __forceinline__ v8f wmma_bf16(v16bf a, v16bf b, v8f c) {
  // (neg_a, A, neg_b, B, c_mod, C, reuse_a, reuse_b)
  return __builtin_amdgcn_wmma_f32_16x16x32_bf16(false, a, false, b,
                                                 (short)0, c, false, false);
}

__device__ __forceinline__ float redmax16(float v) {
  v = fmaxf(v, __shfl_xor(v, 1));
  v = fmaxf(v, __shfl_xor(v, 2));
  v = fmaxf(v, __shfl_xor(v, 4));
  v = fmaxf(v, __shfl_xor(v, 8));
  return v;
}
__device__ __forceinline__ float redsum16(float v) {
  v += __shfl_xor(v, 1);
  v += __shfl_xor(v, 2);
  v += __shfl_xor(v, 4);
  v += __shfl_xor(v, 8);
  return v;
}

// ---------------------------------------------------------------------------
// Tiled GEMM: C[M][N] = A[M][K] * B[N][K]^T   (B row-major [N][K], i.e. W[e][d])
// A: float or bf16 (converted to bf16 in LDS); B: float; C: float or bf16.
// BM=BN=128, BK=32. 256 threads = 8 waves; wave -> 32(M) x 64(N) region.
// ---------------------------------------------------------------------------
template <typename AT, typename CT>
__global__ __launch_bounds__(256) void gemm_wmma_kernel(
    const AT* __restrict__ A, const float* __restrict__ B, CT* __restrict__ C,
    int M, int N, int K) {
  __shared__ bf16_t As[128][40];  // 40 = 32 + pad (16B aligned rows, no bank conflicts)
  __shared__ bf16_t Bs[128][40];

  const int tid = threadIdx.x;
  const int lane = tid & 31;
  const int w = tid >> 5;
  const int lane16 = lane & 15;
  const int lhalf = lane >> 4;
  const int wm = w & 3;   // 4 waves over M (32 rows each)
  const int wn = w >> 2;  // 2 waves over N (64 cols each)
  const int m0 = blockIdx.y * 128;
  const int n0 = blockIdx.x * 128;

  v8f acc[2][4];
#pragma unroll
  for (int a = 0; a < 2; ++a)
#pragma unroll
    for (int b = 0; b < 4; ++b)
#pragma unroll
      for (int j = 0; j < 8; ++j) acc[a][b][j] = 0.0f;

  for (int k0 = 0; k0 < K; k0 += 32) {
    // ---- stage A tile (128 x 32) as bf16 ----
    if (sizeof(AT) == 4) {  // fp32 source: convert
#pragma unroll
      for (int it = 0; it < 4; ++it) {
        const int p = tid + it * 256;        // 0..1023 float4 units
        const int row = p >> 3, c4 = p & 7;  // 8 float4 per row
        const float4 v =
            *(const float4*)((const float*)A + (size_t)(m0 + row) * K + k0 + c4 * 4);
        bf16_t* d = &As[row][c4 * 4];
        d[0] = (bf16_t)v.x; d[1] = (bf16_t)v.y;
        d[2] = (bf16_t)v.z; d[3] = (bf16_t)v.w;
      }
    } else {  // bf16 source: straight 16B copies
#pragma unroll
      for (int it = 0; it < 2; ++it) {
        const int u = tid + it * 256;        // 0..511 units of 8 bf16
        const int row = u >> 2, c8 = u & 3;
        *(v8bf*)&As[row][c8 * 8] =
            *(const v8bf*)((const bf16_t*)A + (size_t)(m0 + row) * K + k0 + c8 * 8);
      }
    }
    // ---- stage B tile (128 x 32), always fp32 source ----
#pragma unroll
    for (int it = 0; it < 4; ++it) {
      const int p = tid + it * 256;
      const int row = p >> 3, c4 = p & 7;
      const float4 v = *(const float4*)(B + (size_t)(n0 + row) * K + k0 + c4 * 4);
      bf16_t* d = &Bs[row][c4 * 4];
      d[0] = (bf16_t)v.x; d[1] = (bf16_t)v.y;
      d[2] = (bf16_t)v.z; d[3] = (bf16_t)v.w;
    }
    // ---- prefetch next K-tile (global_prefetch_b8) while we compute ----
    if (k0 + 32 < K) {
      const int row = tid >> 1, half64 = (tid & 1) * 16;
      __builtin_prefetch((const float*)A + (size_t)(m0 + row) * K + k0 + 32 + half64, 0, 1);
      __builtin_prefetch(B + (size_t)(n0 + row) * K + k0 + 32 + half64, 0, 1);
    }
    __syncthreads();

    // ---- fragments + WMMA ----
    v16bf af[2], bfm[4];
#pragma unroll
    for (int tm = 0; tm < 2; ++tm) {
      const int m = wm * 32 + tm * 16 + lane16;
      // A layout: lane half 0 holds K 0-7 & 16-23, half 1 holds K 8-15 & 24-31
      af[tm] = combine16(*(const v8bf*)&As[m][lhalf * 8],
                         *(const v8bf*)&As[m][16 + lhalf * 8]);
    }
#pragma unroll
    for (int tn = 0; tn < 4; ++tn) {
      const int n = wn * 64 + tn * 16 + lane16;
      // B layout: lane half 0 holds K 0-7 & 16-23, half 1 holds K 8-15 & 24-31
      bfm[tn] = combine16(*(const v8bf*)&Bs[n][lhalf * 8],
                          *(const v8bf*)&Bs[n][16 + lhalf * 8]);
    }
#pragma unroll
    for (int tm = 0; tm < 2; ++tm)
#pragma unroll
      for (int tn = 0; tn < 4; ++tn)
        acc[tm][tn] = wmma_bf16(af[tm], bfm[tn], acc[tm][tn]);
    __syncthreads();
  }

  // ---- store: C/D layout: lane16 = n, vgpr j -> m = lhalf*8 + j ----
#pragma unroll
  for (int tm = 0; tm < 2; ++tm)
#pragma unroll
    for (int tn = 0; tn < 4; ++tn) {
      const int mrow = m0 + wm * 32 + tm * 16 + lhalf * 8;
      const int ncol = n0 + wn * 64 + tn * 16 + lane16;
#pragma unroll
      for (int j = 0; j < 8; ++j)
        C[(size_t)(mrow + j) * N + ncol] = (CT)acc[tm][tn][j];
    }
}

// ---------------------------------------------------------------------------
// V transpose: per (b,h), V[k][d] (bf16, d-fastest) -> Vt[d][k] (k-fastest),
// so attention PV B-operand fragments become contiguous 16B global loads.
// One 32x32 tile per block through padded LDS.
// ---------------------------------------------------------------------------
__global__ __launch_bounds__(256) void transpose_v_kernel(
    const bf16_t* __restrict__ qkv, bf16_t* __restrict__ Vt) {
  constexpr int L = 2048, DH = 128, NH = 16;
  __shared__ bf16_t T[32][36];
  const int tid = threadIdx.x;
  const int k0 = blockIdx.x * 32;
  const int d0 = blockIdx.y * 32;
  const int bh = blockIdx.z;  // b*16 + h
  const int b = bh >> 4, h = bh & 15;
  const bf16_t* Vp = qkv + ((size_t)b * 3 * NH + 2 * NH + h) * L * DH;
  bf16_t* Vo = Vt + (size_t)bh * DH * L;
  {
    const int k = tid >> 3, d4 = (tid & 7) * 4;  // 32 rows x 8 quads
    const bf16_t* src = Vp + (size_t)(k0 + k) * DH + d0 + d4;
    T[k][d4 + 0] = src[0]; T[k][d4 + 1] = src[1];
    T[k][d4 + 2] = src[2]; T[k][d4 + 3] = src[3];
  }
  __syncthreads();
  {
    const int d = tid >> 3, k4 = (tid & 7) * 4;
    bf16_t* dst = Vo + (size_t)(d0 + d) * L + k0 + k4;
    dst[0] = T[k4 + 0][d]; dst[1] = T[k4 + 1][d];
    dst[2] = T[k4 + 2][d]; dst[3] = T[k4 + 3][d];
  }
}

// ---------------------------------------------------------------------------
// Flash attention. qkv is the raw (b, L, 6144) bf16 GEMM output, which is
// byte-identical to row-major [b][3][16][2048][128] (since 6144 = 48*128 the
// reference's reshape(b,3,H,L,D_H) of the (b,L,6144) tensor lands exactly on
// this layout). Vt is pre-transposed V. 128 threads = 4 independent waves;
// each wave owns TWO 16-query tiles (32 queries) so every K / Vt fragment
// loaded from L2 feeds two WMMAs. No barriers in the main loop: K and Vt
// fragments stream from global (L2-resident), P restaging is same-wave LDS
// (hardware-ordered). Output AO[b][q][h*128+d] (bf16).
// ---------------------------------------------------------------------------
__global__ __launch_bounds__(128) void attn_kernel(const bf16_t* __restrict__ qkv,
                                                   const bf16_t* __restrict__ Vt,
                                                   bf16_t* __restrict__ AO) {
  constexpr int L = 2048, DH = 128, NH = 16;
  __shared__ bf16_t Ps[4][2][16][40];  // per-wave, per-qtile P staging [m][k]

  const int tid = threadIdx.x;
  const int lane = tid & 31;
  const int w = tid >> 5;
  const int lane16 = lane & 15;
  const int lhalf = lane >> 4;
  const int qb = blockIdx.x, h = blockIdx.y, b = blockIdx.z;

  const size_t bbase = (size_t)b * 3 * NH * L * DH;
  const bf16_t* Qp = qkv + bbase + (size_t)(0 * NH + h) * L * DH;
  const bf16_t* Kp = qkv + bbase + (size_t)(1 * NH + h) * L * DH;
  const bf16_t* Vtp = Vt + (size_t)(b * NH + h) * DH * L;
  const int q0 = qb * 128 + w * 32;  // 32 queries per wave: tiles q0, q0+16

  // Q fragments: 2 q-tiles x 4 chunks of 16x32 over d
  v16bf qf[2][4];
#pragma unroll
  for (int qt = 0; qt < 2; ++qt)
#pragma unroll
    for (int c = 0; c < 4; ++c) {
      const bf16_t* qr =
          Qp + (size_t)(q0 + qt * 16 + lane16) * DH + c * 32 + lhalf * 8;
      qf[qt][c] = combine16(*(const v8bf*)qr, *(const v8bf*)(qr + 16));
    }

  v8f oacc[2][8];
#pragma unroll
  for (int qt = 0; qt < 2; ++qt)
#pragma unroll
    for (int c = 0; c < 8; ++c)
#pragma unroll
      for (int j = 0; j < 8; ++j) oacc[qt][c][j] = 0.0f;
  float Mr[2][8], Lr[2][8];
#pragma unroll
  for (int qt = 0; qt < 2; ++qt)
#pragma unroll
    for (int j = 0; j < 8; ++j) { Mr[qt][j] = -3.0e38f; Lr[qt][j] = 0.0f; }
  const float cexp = 1.4426950408889634f * 0.08838834764831843f;  // log2e/sqrt(128)

  for (int kb = 0; kb < L / 32; ++kb) {
    // prefetch next key block (K rows + Vt rows)
    if (kb + 1 < L / 32) {
      __builtin_prefetch(Kp + (size_t)((kb + 1) * 32 + lane) * DH, 0, 1);
      __builtin_prefetch(Vtp + (size_t)(lane * 4) * L + (kb + 1) * 32, 0, 1);
    }

    // S = Q K^T for 32 keys: two 16x16 tiles per q-tile.
    // Each K fragment is loaded once and feeds both q-tiles.
    v8f sacc[2][2];
#pragma unroll
    for (int t = 0; t < 2; ++t) {
#pragma unroll
      for (int qt = 0; qt < 2; ++qt)
#pragma unroll
        for (int j = 0; j < 8; ++j) sacc[qt][t][j] = 0.0f;
#pragma unroll
      for (int c = 0; c < 4; ++c) {
        const bf16_t* kr =
            Kp + (size_t)(kb * 32 + t * 16 + lane16) * DH + c * 32 + lhalf * 8;
        const v16bf kf = combine16(*(const v8bf*)kr, *(const v8bf*)(kr + 16));
#pragma unroll
        for (int qt = 0; qt < 2; ++qt)
          sacc[qt][t] = wmma_bf16(qf[qt][c], kf, sacc[qt][t]);
      }
    }

    // online softmax + P staging per q-tile (row m = lhalf*8 + j)
    v16bf pf[2];
#pragma unroll
    for (int qt = 0; qt < 2; ++qt) {
      float p0[8], p1[8], alpha[8];
#pragma unroll
      for (int j = 0; j < 8; ++j) {
        float mx = redmax16(fmaxf(sacc[qt][0][j], sacc[qt][1][j]));
        const float Mn = fmaxf(Mr[qt][j], mx);
        alpha[j] = exp2f((Mr[qt][j] - Mn) * cexp);
        p0[j] = exp2f((sacc[qt][0][j] - Mn) * cexp);
        p1[j] = exp2f((sacc[qt][1][j] - Mn) * cexp);
        const float rs = redsum16(p0[j] + p1[j]);
        Lr[qt][j] = Lr[qt][j] * alpha[j] + rs;
        Mr[qt][j] = Mn;
      }
#pragma unroll
      for (int c = 0; c < 8; ++c)
#pragma unroll
        for (int j = 0; j < 8; ++j) oacc[qt][c][j] *= alpha[j];

      // stage P (C layout -> A layout) via per-wave LDS; same-wave DS in-order
#pragma unroll
      for (int j = 0; j < 8; ++j) {
        Ps[w][qt][lhalf * 8 + j][lane16] = (bf16_t)p0[j];
        Ps[w][qt][lhalf * 8 + j][16 + lane16] = (bf16_t)p1[j];
      }
      pf[qt] = combine16(*(const v8bf*)&Ps[w][qt][lane16][lhalf * 8],
                         *(const v8bf*)&Ps[w][qt][lane16][16 + lhalf * 8]);
    }

    // O += P V : 8 d-chunks; each Vt fragment feeds both q-tiles
#pragma unroll
    for (int c = 0; c < 8; ++c) {
      const bf16_t* vr = Vtp + (size_t)(c * 16 + lane16) * L + kb * 32 + lhalf * 8;
      const v16bf vf = combine16(*(const v8bf*)vr, *(const v8bf*)(vr + 16));
#pragma unroll
      for (int qt = 0; qt < 2; ++qt)
        oacc[qt][c] = wmma_bf16(pf[qt], vf, oacc[qt][c]);
    }
  }

  // normalize + store AO[b][q][h*128 + d]
#pragma unroll
  for (int qt = 0; qt < 2; ++qt) {
    float inv[8];
#pragma unroll
    for (int j = 0; j < 8; ++j) inv[j] = 1.0f / Lr[qt][j];
#pragma unroll
    for (int c = 0; c < 8; ++c)
#pragma unroll
      for (int j = 0; j < 8; ++j) {
        const int q = q0 + qt * 16 + lhalf * 8 + j;
        AO[((size_t)(b * L + q)) * 2048 + h * 128 + c * 16 + lane16] =
            (bf16_t)(oacc[qt][c][j] * inv[j]);
      }
  }
}

// ---------------------------------------------------------------------------
extern "C" void kernel_launch(void* const* d_in, const int* in_sizes, int n_in,
                              void* d_out, int out_size, void* d_ws, size_t ws_size,
                              hipStream_t stream) {
  (void)in_sizes; (void)n_in; (void)out_size; (void)ws_size;
  const float* x = (const float*)d_in[0];      // [2][2048][2048]
  const float* Wqkv = (const float*)d_in[1];   // [6144][2048]
  const float* Wo = (const float*)d_in[2];     // [2048][2048]
  float* out = (float*)d_out;                  // [2][2048][2048]

  bf16_t* qkv = (bf16_t*)d_ws;                        // 2*3*16*2048*128 bf16 = 48 MB
  bf16_t* ao = qkv + (size_t)2 * 3 * 16 * 2048 * 128; // 2*2048*2048 bf16 = 16 MB
  bf16_t* vt = ao + (size_t)2 * 2048 * 2048;          // 2*16*128*2048 bf16 = 16 MB

  // 1) QKV projection: [4096][2048] x [6144][2048]^T -> bf16 [4096][6144]
  gemm_wmma_kernel<float, bf16_t>
      <<<dim3(6144 / 128, 4096 / 128), 256, 0, stream>>>(x, Wqkv, qkv, 4096, 6144, 2048);

  // 2) V transpose per (b,h): [2048][128] -> [128][2048]
  transpose_v_kernel<<<dim3(2048 / 32, 128 / 32, 2 * 16), 256, 0, stream>>>(qkv, vt);

  // 3) flash attention: 128 queries per block (32 per wave)
  attn_kernel<<<dim3(2048 / 128, 16, 2), 128, 0, stream>>>(qkv, vt, ao);

  // 4) output projection: [4096][2048](bf16) x [2048][2048]^T -> f32 out
  gemm_wmma_kernel<bf16_t, float>
      <<<dim3(2048 / 128, 4096 / 128), 256, 0, stream>>>(ao, Wo, out, 4096, 2048, 2048);
}
